// Attention_31928786878759
// MI455X (gfx1250) — compile-verified, matrix operand order checked by reading, not run on hardware
//
#include <hip/hip_runtime.h>
#include <hip/hip_bf16.h>

typedef __attribute__((ext_vector_type(16))) __bf16 v16bf;
typedef __attribute__((ext_vector_type(8)))  float  v8f;
typedef __attribute__((ext_vector_type(4)))  unsigned int v4u;
typedef __attribute__((ext_vector_type(8)))  int    v8i;
typedef __attribute__((ext_vector_type(4)))  int    v4i;

#define SEQ   2048
#define CIN   1024
#define C3    3072
#define HD    64
#define NHEAD 16
#define BATCH 2
#define BN    (BATCH * SEQ)   // 4096 flattened rows

__device__ __forceinline__ int lane_id() { return threadIdx.x & 31; }

__device__ __forceinline__ v8f v8f_zero() {
  v8f z;
#pragma unroll
  for (int i = 0; i < 8; ++i) z[i] = 0.0f;
  return z;
}

// K-index mapping for 16-bit A/B WMMA fragments (ISA 7.12.2):
// lane<16 : halves 0..7 -> K=0..7,  halves 8..15 -> K=16..23
// lane>=16: halves 0..7 -> K=8..15, halves 8..15 -> K=24..31
__device__ __forceinline__ int frag_k(int h, int lane) {
  int kb = (lane < 16) ? 0 : 8;
  return (h < 8) ? (kb + h) : (kb + 8 + h);
}

// A fragment 16x32 bf16: element (m,k) at src[m*mstride + k]
__device__ __forceinline__ v16bf load_frag_a(const __bf16* src, int mstride) {
  int lane = lane_id();
  const __bf16* p = src + (lane & 15) * mstride;
  v16bf r;
#pragma unroll
  for (int h = 0; h < 16; ++h) r[h] = p[frag_k(h, lane)];
  return r;
}

// B fragment 32x16 bf16: element (k,n) at src[k*kstride + n*nstride]
__device__ __forceinline__ v16bf load_frag_b(const __bf16* src, int kstride, int nstride) {
  int lane = lane_id();
  const __bf16* p = src + (lane & 15) * nstride;
  v16bf r;
#pragma unroll
  for (int h = 0; h < 16; ++h) r[h] = p[frag_k(h, lane) * kstride];
  return r;
}

__device__ __forceinline__ v8f wmma_bf16(v16bf a, v16bf b, v8f c) {
  return __builtin_amdgcn_wmma_f32_16x16x32_bf16(false, a, false, b, (short)0, c,
                                                 false, false);
}

// ---------------------------------------------------------------------------
// Tensor Data Mover: load a 32x64 bf16 tile (rows of a [SEQ,64] tensor) into
// LDS with native padding: 32 dwords (one 64-elem row) then +4 dwords pad
// => padded LDS row stride of 72 halves. Tracked on TENSORcnt.
// This toolchain exposes the 6-arg builtin:
//   (uint32x4 g0, int32x8 g1, int32x4 g2, int32x4 g3, int32x8 g4, i32 cpol)
// ---------------------------------------------------------------------------
__device__ __forceinline__ void tdm_load_tile32x64(unsigned lds_byte,
                                                   unsigned long long gaddr) {
  v4u g0;
  g0[0] = 1u;                                        // count=1, user D#
  g0[1] = lds_byte;                                  // lds_addr (bytes)
  g0[2] = (unsigned)(gaddr & 0xffffffffu);           // global_addr[31:0]
  g0[3] = (unsigned)((gaddr >> 32) & 0x01ffffffu)    // global_addr[56:32]
          | 0x80000000u;                             // type=2 ("image")
  v8i g1;
  // data_size=2B(code1), pad_enable, pad_interval=32dw(code4), pad_amount=4dw(code3)
  g1[0] = (int)((1u << 16) | (1u << 20) | (4u << 22) | (3u << 25));
  g1[1] = (int)(64u << 16);            // atomic_barrier=0 | tensor_dim0.lo=64
  g1[2] = (int)((unsigned)SEQ << 16);  // tensor_dim0.hi=0 | tensor_dim1.lo=2048
  g1[3] = (int)(64u << 16);            // tensor_dim1.hi=0 | tile_dim0=64
  g1[4] = 32;                          // tile_dim1=32 | tile_dim2=0
  g1[5] = 64;                          // tensor_dim0_stride = 64 elements
  g1[6] = 0;                           // stride.hi | tensor_dim1_stride.lo
  g1[7] = 0;
  v4i z4;
#pragma unroll
  for (int i = 0; i < 4; ++i) z4[i] = 0;             // 2D: groups 2/3 unused
  v8i z8;
#pragma unroll
  for (int i = 0; i < 8; ++i) z8[i] = 0;
  __builtin_amdgcn_tensor_load_to_lds(g0, g1, z4, z4, z8, 0);
}

// ---------------------------------------------------------------------------
// Kernel 1: qkv = x @ w_qkv^T, fused RoPE on q/k, q pre-scaled by D^-0.5.
// Block = 128 thr (4 waves), computes a 64x64 tile of [BN x 3C].
// fp32 sources need bf16 conversion -> VALU staging + global_prefetch hints.
// ---------------------------------------------------------------------------
__global__ __launch_bounds__(128) void qkv_rope_kernel(
    const float* __restrict__ x, const float* __restrict__ w,
    __bf16* __restrict__ qb, __bf16* __restrict__ kb, __bf16* __restrict__ vb) {
  __shared__ __bf16 lA[64][40];   // x tile  [m][k], padded stride
  __shared__ __bf16 lB[64][40];   // w rows  [n][k]

  const int m0 = blockIdx.x * 64;
  const int n0 = blockIdx.y * 64;
  const int wave = threadIdx.x >> 5;
  const int lane = threadIdx.x & 31;

  v8f acc[4];
#pragma unroll
  for (int t = 0; t < 4; ++t) acc[t] = v8f_zero();

  for (int k0 = 0; k0 < CIN; k0 += 32) {
    if (k0 + 32 < CIN) {   // prefetch next K-chunk (global_prefetch_b8)
      int pr = threadIdx.x >> 1;
      __builtin_prefetch(&x[(size_t)(m0 + pr) * CIN + k0 + 32], 0, 1);
      __builtin_prefetch(&w[(size_t)(n0 + pr) * CIN + k0 + 32], 0, 1);
    }
#pragma unroll 4
    for (int idx = threadIdx.x; idx < 64 * 32; idx += 128) {
      int r = idx >> 5, c = idx & 31;
      lA[r][c] = (__bf16)x[(size_t)(m0 + r) * CIN + k0 + c];
      lB[r][c] = (__bf16)w[(size_t)(n0 + r) * CIN + k0 + c];
    }
    __syncthreads();
    v16bf af = load_frag_a(&lA[wave * 16][0], 40);
#pragma unroll
    for (int t = 0; t < 4; ++t) {
      v16bf bf = load_frag_b(&lB[t * 16][0], 1, 40);   // (k,n) -> lB[n][k]
      acc[t] = wmma_bf16(af, bf, acc[t]);
    }
    __syncthreads();
  }

  const int which = n0 / CIN;            // 0=q 1=k 2=v (tiles never straddle)
  const int head  = (n0 % CIN) / HD;
  const int rbase = wave * 16 + ((lane < 16) ? 0 : 8);

  if (which < 2) {                       // RoPE in-register: pair t <-> t+2
#pragma unroll
    for (int t = 0; t < 2; ++t) {
      int dl = t * 16 + (lane & 15);     // d % 32 (t<2 => d<32)
      float freq = exp2f(-11.0f * (float)dl * (1.0f / 32.0f)); // 2048^(-dl/32)
#pragma unroll
      for (int i = 0; i < 8; ++i) {
        int row = m0 + rbase + i;
        int n = row & (SEQ - 1);
        float ang = (float)n * freq;
        float cv = __cosf(ang), sv = __sinf(ang);
        float a0 = acc[t][i], a2 = acc[t + 2][i];
        acc[t][i]     = a0 * cv - a2 * sv;   // d < 32: x1*cos - x2*sin
        acc[t + 2][i] = a2 * cv + a0 * sv;   // d >=32: x2*cos + x1*sin
      }
    }
  }

  const float qscale = (which == 0) ? 0.125f : 1.0f;   // D^-0.5
  __bf16* dst = (which == 0) ? qb : (which == 1) ? kb : vb;
#pragma unroll
  for (int t = 0; t < 4; ++t) {
    int d = t * 16 + (lane & 15);
#pragma unroll
    for (int i = 0; i < 8; ++i) {
      int row = m0 + rbase + i;
      int b = row >> 11;                 // / SEQ
      int n = row & (SEQ - 1);
      dst[(((size_t)b * NHEAD + head) * SEQ + n) * HD + d] =
          (__bf16)(acc[t][i] * qscale);
    }
  }
}

// ---------------------------------------------------------------------------
// Kernel 2: flash attention per (b,h). Block = 4 waves x 16 queries = 64 q.
// K/V chunks (32 keys) staged by the Tensor Data Mover, double-buffered so
// the chunk-(i+1) DMA overlaps chunk-i WMMA work.
// ---------------------------------------------------------------------------
__global__ __launch_bounds__(128) void attn_kernel(
    const __bf16* __restrict__ Q, const __bf16* __restrict__ K,
    const __bf16* __restrict__ V, const int* __restrict__ mask,
    __bf16* __restrict__ O) {
  __shared__ __bf16 lK[2][32][72];       // [buf][key][d], TDM-padded rows
  __shared__ __bf16 lV[2][32][72];
  __shared__ __bf16 lP[4][16][40];       // per-wave P tile [q][key]

  const int bh = blockIdx.x;             // 0..31
  const int b = bh >> 4, h = bh & 15;
  const int wave = threadIdx.x >> 5;
  const int lane = threadIdx.x & 31;
  const int q0 = blockIdx.y * 64 + wave * 16;

  const __bf16* Qp = Q + (size_t)bh * SEQ * HD;
  const __bf16* Kp = K + (size_t)bh * SEQ * HD;
  const __bf16* Vp = V + (size_t)bh * SEQ * HD;
  const int* mp = mask + b * SEQ;

  const v16bf qa0 = load_frag_a(Qp + (size_t)q0 * HD + 0, HD);   // d 0..31
  const v16bf qa1 = load_frag_a(Qp + (size_t)q0 * HD + 32, HD);  // d 32..63

  float mi[8], li[8];
  v8f acc[4];
#pragma unroll
  for (int i = 0; i < 8; ++i) { mi[i] = -3.0e38f; li[i] = 0.0f; }
#pragma unroll
  for (int t = 0; t < 4; ++t) acc[t] = v8f_zero();

  // preload chunk 0 into buffer 0 via TDM (wave 0 issues; EXEC is ignored)
  if (wave == 0) {
    tdm_load_tile32x64((unsigned)(size_t)&lK[0][0][0],
                       (unsigned long long)(size_t)Kp);
    tdm_load_tile32x64((unsigned)(size_t)&lV[0][0][0],
                       (unsigned long long)(size_t)Vp);
  }

  for (int kchunk = 0; kchunk < SEQ; kchunk += 32) {
    const int buf = (kchunk >> 5) & 1;
    __syncthreads();                     // prior reads of buf^1 complete
    if (wave == 0) {
      if (kchunk + 32 < SEQ) {           // kick off next chunk's DMA first
        tdm_load_tile32x64((unsigned)(size_t)&lK[buf ^ 1][0][0],
                           (unsigned long long)(size_t)(Kp + (size_t)(kchunk + 32) * HD));
        tdm_load_tile32x64((unsigned)(size_t)&lV[buf ^ 1][0][0],
                           (unsigned long long)(size_t)(Vp + (size_t)(kchunk + 32) * HD));
        __builtin_amdgcn_s_wait_tensorcnt(2);   // current chunk landed
      } else {
        __builtin_amdgcn_s_wait_tensorcnt(0);
      }
    }
    __syncthreads();                     // current buf visible to all waves

    // ---- S = Q K^T (scale already folded into Q), + key padding mask ----
    v8f s[2];
#pragma unroll
    for (int sub = 0; sub < 2; ++sub) {
      v16bf b0 = load_frag_b(&lK[buf][sub * 16][0], 1, 72);   // (d,key) d 0..31
      v16bf b1 = load_frag_b(&lK[buf][sub * 16][32], 1, 72);  // d 32..63
      v8f sv = v8f_zero();
      sv = wmma_bf16(qa0, b0, sv);
      sv = wmma_bf16(qa1, b1, sv);
      int key = kchunk + sub * 16 + (lane & 15);
      float madd = (mp[key] == 0) ? -1.0e9f : 0.0f;
#pragma unroll
      for (int i = 0; i < 8; ++i) sv[i] += madd;
      s[sub] = sv;
    }

    // ---- online softmax (row m lives at vgpr i within a 16-lane half) ----
#pragma unroll
    for (int i = 0; i < 8; ++i) {
      float rm = fmaxf(s[0][i], s[1][i]);
#pragma unroll
      for (int off = 8; off > 0; off >>= 1)
        rm = fmaxf(rm, __shfl_xor(rm, off, 32));
      float mnew = fmaxf(mi[i], rm);
      float corr = __expf(mi[i] - mnew);
      float p0 = __expf(s[0][i] - mnew);
      float p1 = __expf(s[1][i] - mnew);
      s[0][i] = p0; s[1][i] = p1;
      float ps = p0 + p1;
#pragma unroll
      for (int off = 8; off > 0; off >>= 1)
        ps += __shfl_xor(ps, off, 32);
      li[i] = li[i] * corr + ps;
      mi[i] = mnew;
#pragma unroll
      for (int t = 0; t < 4; ++t) acc[t][i] *= corr;
    }

    // ---- P through per-wave LDS to become an A fragment ----
    int rb = (lane < 16) ? 0 : 8;
#pragma unroll
    for (int i = 0; i < 8; ++i) {
      lP[wave][rb + i][(lane & 15)]      = (__bf16)s[0][i];
      lP[wave][rb + i][16 + (lane & 15)] = (__bf16)s[1][i];
    }
    asm volatile("s_wait_dscnt 0x0" ::: "memory");

    v16bf pf = load_frag_a(&lP[wave][0][0], 40);
#pragma unroll
    for (int t = 0; t < 4; ++t) {
      v16bf vf = load_frag_b(&lV[buf][0][t * 16], 72, 1);     // (key, d)
      acc[t] = wmma_bf16(pf, vf, acc[t]);
    }
  }

  // ---- normalize, store O as bf16 in [B, N, C] for the final GEMM ----
  int rb = (lane < 16) ? 0 : 8;
#pragma unroll
  for (int t = 0; t < 4; ++t) {
#pragma unroll
    for (int i = 0; i < 8; ++i) {
      int row = q0 + rb + i;
      float o = acc[t][i] / li[i];
      O[((size_t)(b * SEQ + row)) * CIN + h * HD + t * 16 + (lane & 15)] =
          (__bf16)o;
    }
  }
}

// ---------------------------------------------------------------------------
// Kernel 3: out = O @ w_proj^T + b_proj  (fp32 output).
// A tile is already bf16 in memory -> stage it with per-lane async
// global->LDS B128 copies (ASYNCcnt path); w_proj needs fp32->bf16 convert.
// ---------------------------------------------------------------------------
__global__ __launch_bounds__(128) void proj_kernel(
    const __bf16* __restrict__ A, const float* __restrict__ w,
    const float* __restrict__ bias, float* __restrict__ out) {
  __shared__ __bf16 lA[64][40];          // row stride 80 bytes (16B-aligned)
  __shared__ __bf16 lB[64][40];

  const int m0 = blockIdx.x * 64;
  const int n0 = blockIdx.y * 64;
  const int wave = threadIdx.x >> 5;
  const int lane = threadIdx.x & 31;
  const unsigned ldsA = (unsigned)(size_t)&lA[0][0];

  v8f acc[4];
#pragma unroll
  for (int t = 0; t < 4; ++t) acc[t] = v8f_zero();

  for (int k0 = 0; k0 < CIN; k0 += 32) {
    // async copy A tile: 64 rows x 64B, as 16B chunks, 2 per lane
#pragma unroll
    for (int j = 0; j < 2; ++j) {
      int idx = threadIdx.x + j * 128;   // 0..255
      int r = idx >> 2, c4 = idx & 3;
      unsigned long long ga =
          (unsigned long long)(size_t)(A + (size_t)(m0 + r) * CIN + k0 + c4 * 8);
      unsigned la = ldsA + (unsigned)(r * 80 + c4 * 16);
      asm volatile("global_load_async_to_lds_b128 %0, %1, off"
                   :: "v"(la), "v"(ga) : "memory");
    }
#pragma unroll 4
    for (int idx = threadIdx.x; idx < 64 * 32; idx += 128) {
      int r = idx >> 5, c = idx & 31;
      lB[r][c] = (__bf16)w[(size_t)(n0 + r) * CIN + k0 + c];
    }
    asm volatile("s_wait_asynccnt 0x0" ::: "memory");
    __syncthreads();
    v16bf af = load_frag_a(&lA[wave * 16][0], 40);
#pragma unroll
    for (int t = 0; t < 4; ++t) {
      v16bf bf = load_frag_b(&lB[t * 16][0], 1, 40);
      acc[t] = wmma_bf16(af, bf, acc[t]);
    }
    __syncthreads();
  }

  const int rbase = wave * 16 + ((lane < 16) ? 0 : 8);
#pragma unroll
  for (int t = 0; t < 4; ++t) {
    int col = n0 + t * 16 + (lane & 15);
    float bv = bias[col];
#pragma unroll
    for (int i = 0; i < 8; ++i) {
      int row = m0 + rbase + i;
      out[(size_t)row * CIN + col] = acc[t][i] + bv;
    }
  }
}

// ---------------------------------------------------------------------------
extern "C" void kernel_launch(void* const* d_in, const int* in_sizes, int n_in,
                              void* d_out, int out_size, void* d_ws, size_t ws_size,
                              hipStream_t stream) {
  const float* x      = (const float*)d_in[0];
  const float* w_qkv  = (const float*)d_in[1];
  const float* w_proj = (const float*)d_in[2];
  const float* b_proj = (const float*)d_in[3];
  const int*   mask   = (const int*)d_in[4];
  float* out = (float*)d_out;

  const size_t QKV_ELEMS = (size_t)BATCH * NHEAD * SEQ * HD;   // 4M bf16 each
  __bf16* wsQ = (__bf16*)d_ws;
  __bf16* wsK = wsQ + QKV_ELEMS;
  __bf16* wsV = wsK + QKV_ELEMS;
  __bf16* wsO = wsV + QKV_ELEMS;                               // [BN, C]

  qkv_rope_kernel<<<dim3(BN / 64, C3 / 64), 128, 0, stream>>>(
      x, w_qkv, wsQ, wsK, wsV);
  attn_kernel<<<dim3(BATCH * NHEAD, SEQ / 64), 128, 0, stream>>>(
      wsQ, wsK, wsV, mask, wsO);
  proj_kernel<<<dim3(BN / 64, CIN / 64), 128, 0, stream>>>(
      wsO, w_proj, b_proj, out);
}
